// Head_66331474920119
// MI455X (gfx1250) — matrix operand order repaired and checked by reading, not compiled, not run on hardware
//
#include <hip/hip_runtime.h>
#include <math.h>

// ---------------------------------------------------------------------------
// Single-head causal attention, CDNA5 (gfx1250, wave32, WMMA bf16).
// B=4, T=4096, C=2048, H=128.
//   q = x@Wq ; k = x@Wk ; v = x@Wv            (bf16 WMMA, f32 accum)
//   wei = softmax(mask(q k^T * 2048^-0.05))   (flash-attention, online softmax)
//   out = wei @ v
// Attention kernel double-buffers K/V tiles in LDS via GLOBAL_LOAD_ASYNC_TO_LDS
// (ASYNCcnt), overlapping the async DMA with WMMA + softmax of the previous tile.
// ---------------------------------------------------------------------------

typedef __attribute__((ext_vector_type(16))) __bf16 v16bf;
typedef __attribute__((ext_vector_type(8)))  __bf16 v8bf;
typedef __attribute__((ext_vector_type(8)))  float  v8f;
typedef __attribute__((ext_vector_type(4)))  float  v4f;

#define B_   4
#define T_   4096
#define C_   2048
#define H_   128
#define BT_  (B_ * T_)          // 16384 rows
#define NTILES_ (BT_ / 16)      // 1024 16-row tiles
#define WAVES_PER_BLOCK 4

// 2048^(-0.05) = exp(-0.05 * ln 2048)
#define K_SCALE 0.6830190f

// workspace byte offsets
#define QB_OFF  ((size_t)0)
#define KB_OFF  ((size_t)BT_ * H_ * 2)            // 4 MiB
#define VT_OFF  ((size_t)2 * BT_ * H_ * 2)        // 8 MiB   v stored [H][BT]
#define WT_OFF  ((size_t)3 * BT_ * H_ * 2)        // 12 MiB  Wt bf16 [3][H][C]

static __device__ inline v8f wmma_bf16(v16bf a, v16bf b, v8f c) {
  // D = A(16x32) * B(32x16) + C(16x16), f32 accum
  return __builtin_amdgcn_wmma_f32_16x16x32_bf16(
      false, a, false, b, (short)0, c, false, false);
}

static __device__ inline v16bf cat8(v8bf lo, v8bf hi) {
  return __builtin_shufflevector(lo, hi, 0, 1, 2, 3, 4, 5, 6, 7,
                                 8, 9, 10, 11, 12, 13, 14, 15);
}

// 16-byte async copy global -> LDS (ASYNCcnt). lds_off = low 32 bits of the
// generic pointer into __shared__ (== LDS byte offset on AMDGPU).
static __device__ inline void async_ld16(const void* g, unsigned lds_off) {
  asm volatile("global_load_async_to_lds_b128 %0, %1, off"
               :: "v"(lds_off), "v"(g) : "memory");
}

// ---------------------------------------------------------------------------
// Kernel 0: Wq/Wk/Wv [C][H] f32  ->  Wt [3][H][C] bf16 (transposed)
// ---------------------------------------------------------------------------
__global__ __launch_bounds__(256) void wconv_kernel(
    const float* __restrict__ Wq, const float* __restrict__ Wk,
    const float* __restrict__ Wv, __bf16* __restrict__ wt) {
  int i = blockIdx.x * 256 + threadIdx.x;       // 0 .. 3*H*C-1
  int m = i / (H_ * C_);
  int j = i % (H_ * C_);
  int h = j / C_;
  int c = j % C_;
  const float* W = (m == 0) ? Wq : ((m == 1) ? Wk : Wv);
  wt[(size_t)m * H_ * C_ + (size_t)h * C_ + c] =
      (__bf16)W[(size_t)c * H_ + h];
}

// ---------------------------------------------------------------------------
// Kernel 1: projections. grid.y selects {q,k,v}. Each wave: 16 rows x 128 h.
//   A-frag (x, f32->bf16): lane l -> row l%16, K = 8*(l/16)+0..7, 16+8*(l/16)+0..7
//   B-frag (Wt bf16):      lane l -> col l%16, K = 8*(l/16)+0..7, 16+8*(l/16)+0..7
// ---------------------------------------------------------------------------
__global__ __launch_bounds__(128) void proj_kernel(
    const float* __restrict__ x, const __bf16* __restrict__ wt,
    __bf16* __restrict__ qb, __bf16* __restrict__ kb,
    __bf16* __restrict__ vtb) {
  const int lane = threadIdx.x & 31;
  const int wid  = threadIdx.x >> 5;
  const int mt   = blockIdx.x * WAVES_PER_BLOCK + wid;  // 16-row tile id
  const int mat  = blockIdx.y;                          // 0=q 1=k 2=v
  const int n    = lane & 15;
  const int half = lane >> 4;
  const int row0 = mt * 16;

  const __bf16* w = wt + (size_t)mat * H_ * C_;
  const float* xrow = x + (size_t)(row0 + n) * C_;      // A: lane row = l%16
  const int kbase = half * 8;

  v8f acc[8];
#pragma unroll
  for (int t = 0; t < 8; ++t) acc[t] = (v8f)(0.0f);

  for (int kc = 0; kc < C_; kc += 32) {
    // ---- A fragment: two 8xf32 b128-pair loads, convert to bf16
    v4f a0 = *(const v4f*)(xrow + kc + kbase);
    v4f a1 = *(const v4f*)(xrow + kc + kbase + 4);
    v4f a2 = *(const v4f*)(xrow + kc + 16 + kbase);
    v4f a3 = *(const v4f*)(xrow + kc + 16 + kbase + 4);
    v16bf af;
#pragma unroll
    for (int j = 0; j < 4; ++j) {
      af[j]      = (__bf16)a0[j];
      af[4 + j]  = (__bf16)a1[j];
      af[8 + j]  = (__bf16)a2[j];
      af[12 + j] = (__bf16)a3[j];
    }
    if (kc + 32 < C_)
      __builtin_prefetch(xrow + kc + 32 + kbase, 0, 1);  // global_prefetch_b8

#pragma unroll
    for (int nt = 0; nt < 8; ++nt) {
      const __bf16* wr = w + (size_t)(nt * 16 + n) * C_ + kc + 8 * half;
      v16bf bf = cat8(*(const v8bf*)wr, *(const v8bf*)(wr + 16));
      acc[nt] = wmma_bf16(af, bf, acc[nt]);
    }
  }

  // ---- store: C/D layout -> lane (n,half), vgpr r -> row = r + 8*half
  if (mat == 2) {
    // v transposed: vtb[h][t]
#pragma unroll
    for (int nt = 0; nt < 8; ++nt) {
#pragma unroll
      for (int r = 0; r < 8; ++r) {
        int t = row0 + r + 8 * half;
        vtb[(size_t)(nt * 16 + n) * BT_ + t] = (__bf16)acc[nt][r];
      }
    }
  } else {
    __bf16* dst = (mat == 0) ? qb : kb;
#pragma unroll
    for (int nt = 0; nt < 8; ++nt) {
#pragma unroll
      for (int r = 0; r < 8; ++r) {
        int t = row0 + r + 8 * half;
        dst[(size_t)t * H_ + nt * 16 + n] = (__bf16)acc[nt][r];
      }
    }
  }
}

// ---------------------------------------------------------------------------
// Kernel 2: flash attention. Block = 64 query rows (4 waves x 16).
// 32-key K/V tiles double-buffered in LDS via async loads; one barrier per
// tile; uniform trip count per block (causal mask handles the ragged edge).
// ---------------------------------------------------------------------------
__global__ __launch_bounds__(128) void attn_kernel(
    const __bf16* __restrict__ qb, const __bf16* __restrict__ kb,
    const __bf16* __restrict__ vtb, float* __restrict__ out) {
  __shared__ __bf16 kt_lds[2][32 * H_];                // [buf][key][h]  16KB
  __shared__ __bf16 vt_lds[2][H_ * 32];                // [buf][h][key]  16KB
  __shared__ __bf16 p_lds[WAVES_PER_BLOCK][16 * 32];   // per-wave P      4KB

  const int tid  = threadIdx.x;
  const int lane = tid & 31;
  const int wid  = tid >> 5;
  const int n    = lane & 15;
  const int half = lane >> 4;
  const int kbase = half * 8;

  const int q0_blk  = blockIdx.x * 64;        // flat over B*T (64 | 4096)
  const int b       = q0_blk / T_;
  const int lq0_blk = q0_blk - b * T_;
  const int q0  = q0_blk + wid * 16;          // this wave's 16 query rows
  const int lq0 = lq0_blk + wid * 16;
  __bf16* pl = &p_lds[wid][0];

  // ---- load Q A-fragments once: 4 chunks of K=32 over H=128
  v16bf qf[4];
  {
    const __bf16* qrow = qb + (size_t)(q0 + n) * H_;
#pragma unroll
    for (int kc = 0; kc < 4; ++kc) {
      v8bf lo = *(const v8bf*)(qrow + kc * 32 + kbase);
      v8bf hi = *(const v8bf*)(qrow + kc * 32 + 16 + kbase);
      qf[kc] = cat8(lo, hi);
    }
  }

  v8f o[8];
#pragma unroll
  for (int t = 0; t < 8; ++t) o[t] = (v8f)(0.0f);
  float m_i[8], l_i[8];
#pragma unroll
  for (int r = 0; r < 8; ++r) { m_i[r] = -3.0e38f; l_i[r] = 0.0f; }

  // cover keys 0 .. lq0_blk+63 with 32-key tiles (uniform across block)
  const int ktiles = (lq0_blk + 95) >> 5;
  const int kbatch = b * T_;                  // flat key base of this batch

  // ---- stage helper (8x16B async chunks per thread per tile)
  auto stage = [&](int jt, int buf) {
    const int t0g = kbatch + jt * 32;
    const __bf16* ksrc = kb + (size_t)t0g * H_;        // 8KB contiguous
    __bf16* kdst = &kt_lds[buf][0];
#pragma unroll
    for (int c = 0; c < 4; ++c) {
      int idx = tid + c * 128;                         // 16B chunk id, 0..511
      async_ld16(ksrc + idx * 8, (unsigned)(uintptr_t)(kdst + idx * 8));
    }
    __bf16* vdst = &vt_lds[buf][0];
#pragma unroll
    for (int c = 0; c < 4; ++c) {
      int idx = tid + c * 128;                         // 0..511
      int h = idx >> 2, part = idx & 3;
      async_ld16(vtb + (size_t)h * BT_ + t0g + part * 8,
                 (unsigned)(uintptr_t)(vdst + h * 32 + part * 8));
    }
  };

  stage(0, 0);                                         // prologue

  for (int jt = 0; jt < ktiles; ++jt) {
    const int buf = jt & 1;
    const int t0l = jt * 32;                           // local key base

    // only tile jt's copies are outstanding here
    asm volatile("s_wait_asynccnt 0" ::: "memory");
    __syncthreads();   // publish tile jt; buf^1 reads (iter jt-1) retired

    if (jt + 1 < ktiles) {
      stage(jt + 1, buf ^ 1);                          // overlap with compute
      __builtin_prefetch(kb + (size_t)(kbatch + t0l + 64) * H_, 0, 1);
    }

    // ---- S = Q K^T  (two 16-col sub-tiles) from LDS
    v8f s0 = (v8f)(0.0f), s1 = (v8f)(0.0f);
    {
      const __bf16* k0 = &kt_lds[buf][(size_t)n * H_ + 8 * half];
      const __bf16* k1 = &kt_lds[buf][(size_t)(16 + n) * H_ + 8 * half];
#pragma unroll
      for (int kc = 0; kc < 4; ++kc) {
        v16bf b0 = cat8(*(const v8bf*)(k0 + kc * 32),
                        *(const v8bf*)(k0 + kc * 32 + 16));
        v16bf b1 = cat8(*(const v8bf*)(k1 + kc * 32),
                        *(const v8bf*)(k1 + kc * 32 + 16));
        s0 = wmma_bf16(qf[kc], b0, s0);
        s1 = wmma_bf16(qf[kc], b1, s1);
      }
    }

    // ---- scale, causal mask, online softmax (row M = r + 8*half)
    float corr[8];
#pragma unroll
    for (int r = 0; r < 8; ++r) {
      const int tq = lq0 + r + 8 * half;
      float a0 = s0[r] * K_SCALE;
      float a1 = s1[r] * K_SCALE;
      if (t0l + n > tq)      a0 = -1.0e30f;
      if (t0l + 16 + n > tq) a1 = -1.0e30f;
      float mx = fmaxf(a0, a1);
      mx = fmaxf(mx, __shfl_xor(mx, 1, 32));
      mx = fmaxf(mx, __shfl_xor(mx, 2, 32));
      mx = fmaxf(mx, __shfl_xor(mx, 4, 32));
      mx = fmaxf(mx, __shfl_xor(mx, 8, 32));   // within 16-lane half
      const float nm = fmaxf(m_i[r], mx);
      const float cr = __expf(m_i[r] - nm);
      const float p0 = __expf(a0 - nm);
      const float p1 = __expf(a1 - nm);
      float ps = p0 + p1;
      ps += __shfl_xor(ps, 1, 32);
      ps += __shfl_xor(ps, 2, 32);
      ps += __shfl_xor(ps, 4, 32);
      ps += __shfl_xor(ps, 8, 32);
      l_i[r] = l_i[r] * cr + ps;
      m_i[r] = nm;
      corr[r] = cr;
      // stage P tile (C-layout) into per-wave LDS for A-layout reload
      const int M = r + 8 * half;
      pl[M * 32 + n]      = (__bf16)p0;
      pl[M * 32 + 16 + n] = (__bf16)p1;
    }
#pragma unroll
    for (int nt = 0; nt < 8; ++nt)
#pragma unroll
      for (int r = 0; r < 8; ++r) o[nt][r] *= corr[r];

    // per-wave LDS RAW: DS ops are in-order per wave; wait split counter
    asm volatile("s_wait_dscnt 0" ::: "memory");

    // ---- P A-fragment from LDS (transposed lane ownership)
    v16bf pf;
    {
      const __bf16* pr = pl + n * 32;                 // row m = l%16
      v8bf lo = *(const v8bf*)(pr + kbase);           // K = kbase+0..7
      v8bf hi = *(const v8bf*)(pr + 16 + kbase);      // K = 16+kbase+0..7
      pf = cat8(lo, hi);
    }

    // ---- O += P @ V  (V B-frag from LDS: lane col h = nt*16+n)
#pragma unroll
    for (int nt = 0; nt < 8; ++nt) {
      const __bf16* vr = &vt_lds[buf][(size_t)(nt * 16 + n) * 32 + 8 * half];
      v16bf vf = cat8(*(const v8bf*)vr, *(const v8bf*)(vr + 16));
      o[nt] = wmma_bf16(pf, vf, o[nt]);
    }
  }

  // ---- epilogue: out[q][h] = o / l
#pragma unroll
  for (int nt = 0; nt < 8; ++nt) {
#pragma unroll
    for (int r = 0; r < 8; ++r) {
      const int t = q0 + r + 8 * half;
      out[(size_t)t * H_ + nt * 16 + n] = o[nt][r] / l_i[r];
    }
  }
}

// ---------------------------------------------------------------------------
extern "C" void kernel_launch(void* const* d_in, const int* in_sizes, int n_in,
                              void* d_out, int out_size, void* d_ws,
                              size_t ws_size, hipStream_t stream) {
  (void)in_sizes; (void)n_in; (void)out_size; (void)ws_size;
  const float* x  = (const float*)d_in[0];
  const float* Wq = (const float*)d_in[1];
  const float* Wk = (const float*)d_in[2];
  const float* Wv = (const float*)d_in[3];
  float* out = (float*)d_out;

  char* ws = (char*)d_ws;
  __bf16* qb  = (__bf16*)(ws + QB_OFF);
  __bf16* kb  = (__bf16*)(ws + KB_OFF);
  __bf16* vtb = (__bf16*)(ws + VT_OFF);
  __bf16* wt  = (__bf16*)(ws + WT_OFF);

  // 0: transpose+convert weights (3*H*C elements)
  wconv_kernel<<<dim3((3 * H_ * C_) / 256), dim3(256), 0, stream>>>(
      Wq, Wk, Wv, wt);

  // 1: q/k/v projections (grid.y = matrix)
  proj_kernel<<<dim3(NTILES_ / WAVES_PER_BLOCK, 3), dim3(128), 0, stream>>>(
      x, wt, qb, kb, vtb);

  // 2: flash attention (block = 64 query rows)
  attn_kernel<<<dim3(BT_ / 64), dim3(128), 0, stream>>>(qb, kb, vtb, out);
}
